// LinformerAttention_5119601017045
// MI455X (gfx1250) — matrix operand order repaired and checked by reading, not compiled, not run on hardware
//
#include <hip/hip_runtime.h>

// Linformer attention for gfx1250 (MI455X), bf16 WMMA path, f32 accumulate.
// Roofline: ~0.57 GB mandatory HBM traffic (~24 us @ 23.3 TB/s); ~34 GFLOP of
// GEMM on v_wmma_f32_16x16x32_bf16. The per-(b,h) Pk/PvT slices in the fused
// attention kernel are block-shared, so they are DMA'd into LDS once per block
// with the Tensor Data Mover (tensor_load_to_lds) instead of being re-read by
// all 8 waves.

typedef __attribute__((ext_vector_type(16))) __bf16 v16bf;
typedef __attribute__((ext_vector_type(8)))  __bf16 v8bf;
typedef __attribute__((ext_vector_type(8)))  float  v8f;
typedef __attribute__((ext_vector_type(4)))  unsigned int v4u;
typedef __attribute__((ext_vector_type(8)))  int  v8i;
typedef __attribute__((ext_vector_type(4)))  int  v4i;

#define WMMA_BF16(a, b, c) \
  __builtin_amdgcn_wmma_f32_16x16x32_bf16(false, (a), false, (b), (short)0, (c), false, false)

constexpr int Bc = 8, Hc = 16, Nc = 4096, DHc = 64, Kc = 128, Dc = 1024;

// ---------------------------------------------------------------------------
// TDM 2D tile load: bf16 elements (data_size code 1 = 2 bytes).
// D# per CDNA5 ISA ch.8: group0 = {count=1, lds_addr, global_addr[56:0],
// type=2}; group1 = {data_size, tensor_dim0/1, tile_dim0/1, dim0 stride};
// groups 2/3 zero (<=2D tensor). Completion tracked with TENSORcnt.
// This toolchain exposes the 6-arg builtin form (extra int32x8 group).
// ---------------------------------------------------------------------------
__device__ __forceinline__ void tdm_load_2d_bf16(
    unsigned int lds_byte_addr, unsigned long long gaddr,
    unsigned int tensor_d0, unsigned int tensor_d1,
    unsigned int d0_stride, unsigned int tile_d0, unsigned int tile_d1)
{
  v4u g0;
  g0[0] = 1u;                                        // count=1, user D#
  g0[1] = lds_byte_addr;                             // bits 63:32
  g0[2] = (unsigned int)gaddr;                       // bits 95:64
  g0[3] = (unsigned int)(gaddr >> 32) | (2u << 30);  // addr[56:32] | type=2
  v8i g1;
  g1[0] = (int)(1u << 16);                           // data_size=1 (2B)
  g1[1] = (int)((tensor_d0 & 0xFFFFu) << 16);        // tensor_dim0[15:0]
  g1[2] = (int)((tensor_d0 >> 16) | ((tensor_d1 & 0xFFFFu) << 16));
  g1[3] = (int)((tensor_d1 >> 16) | (tile_d0 << 16));// tile_dim0
  g1[4] = (int)(tile_d1 & 0xFFFFu);                  // tile_dim1, tile_dim2=0
  g1[5] = (int)d0_stride;                            // dim0_stride[31:0]
  g1[6] = 0;                                         // stride[47:32], d1_stride
  g1[7] = 0;
  const v4i z4 = {0, 0, 0, 0};
  const v8i z8 = {0, 0, 0, 0, 0, 0, 0, 0};
  __builtin_amdgcn_tensor_load_to_lds(g0, g1, z4, z4, z8, 0);
}

// ---------------------------------------------------------------------------
// Kernel 1: P[b,k,d] = sum_n proj[n,k] * src[b,n,d]   (bf16 output)
//   TRANS == false : out layout [B, K, D]   (keys_p; d-contiguous)
//   TRANS == true  : out layout [B, D, K]   (vals_p transposed; k-contiguous)
// 256-thread workgroup computes the full [K=128 x 128d] slab for one batch;
// per n-step of 32 it stages projT/src tiles into LDS as bf16 with coalesced
// global reads (each element read exactly once); fragments read back as
// contiguous ds_load_b128 pairs. 8 WMMAs per wave per step.
// ---------------------------------------------------------------------------
template <bool TRANS>
__global__ __launch_bounds__(256) void linformer_proj_kernel(
    const float* __restrict__ src,    // [B, N, D]
    const float* __restrict__ proj,   // [N, K]
    __bf16* __restrict__ out)
{
  __shared__ __bf16 lds_proj[Kc * 32];    // [k][r], 8 KB
  __shared__ __bf16 lds_src [128 * 32];   // [c][r], 8 KB

  const int t    = threadIdx.x;
  const int lane = t & 31;
  const int wave = t >> 5;
  const int d0   = blockIdx.x * 128;
  const int b    = blockIdx.y;
  const int half = lane >> 4;
  const int lm   = lane & 15;
  const int kt   = wave;

  v8f acc[8] = {};

  const int stg_c = t & 127;
  const int stg_r = t >> 7;

  for (int n0 = 0; n0 < Nc; n0 += 32) {
#pragma unroll
    for (int it = 0; it < 16; ++it) {
      const int r = it * 2 + stg_r;
      lds_proj[stg_c * 32 + r] =
          (__bf16)proj[(size_t)(n0 + r) * Kc + stg_c];
      lds_src[stg_c * 32 + r] =
          (__bf16)src[((size_t)b * Nc + n0 + r) * Dc + d0 + stg_c];
    }
    __syncthreads();

    union { v16bf v; v8bf h[2]; } afr;
    const __bf16* arow = &lds_proj[(kt * 16 + lm) * 32];
    afr.h[0] = *(const v8bf*)&arow[half * 16];
    afr.h[1] = *(const v8bf*)&arow[half * 16 + 8];

#pragma unroll
    for (int j = 0; j < 8; ++j) {
      const v16bf bb =
          *(const v16bf*)&lds_src[(j * 16 + lm) * 32 + half * 16];
      acc[j] = WMMA_BF16(afr.v, bb, acc[j]);
    }
    __syncthreads();
  }

#pragma unroll
  for (int j = 0; j < 8; ++j)
#pragma unroll
    for (int i = 0; i < 8; ++i) {
      const int krow = kt * 16 + i + half * 8;
      const int dcol = d0 + j * 16 + lm;
      const __bf16 v = (__bf16)acc[j][i];
      if constexpr (!TRANS)
        out[((size_t)b * Kc + krow) * Dc + dcol] = v;
      else
        out[((size_t)b * Dc + dcol) * Kc + krow] = v;
    }
}

// ---------------------------------------------------------------------------
// Kernel 2: fused attention per (b, h, 128 query rows).
// Wave 0 TDM-loads the (b,h) slices Pk[128 x 64] and PvT[64 x 128] into LDS;
// all 8 waves then compute dots -> softmax -> attn@V from LDS fragments.
// ---------------------------------------------------------------------------
__global__ __launch_bounds__(256) void linformer_attn_kernel(
    const float*  __restrict__ q,    // [B, H, N, DH]
    const __bf16* __restrict__ Pk,   // [B, K, D]
    const __bf16* __restrict__ PvT,  // [B, D, K]
    float* __restrict__ out)         // [B, N, D]
{
  __shared__ __bf16 lds_pk[Kc * DHc];       // [k][dh], 16 KB
  __shared__ __bf16 lds_pv[DHc * Kc];       // [dh][k], 16 KB
  __shared__ __bf16 lds_attn[8][16 * Kc];   // 4 KB per wave, 32 KB

  const int lane = threadIdx.x & 31;
  const int wave = threadIdx.x >> 5;
  const int n0   = blockIdx.x * 128 + wave * 16;
  const int h    = blockIdx.y;
  const int b    = blockIdx.z;
  const int half = lane >> 4;
  const int lm   = lane & 15;

  // --- TDM: stage Pk / PvT head-slices into LDS (one wave issues DMA) ------
  if (wave == 0) {
    // Pk slice: 128 rows (k) x 64 elems (dh), row stride Dc elems.
    tdm_load_2d_bf16(
        (unsigned int)(size_t)&lds_pk[0],
        (unsigned long long)(size_t)(Pk + ((size_t)b * Kc) * Dc + h * DHc),
        Dc, Kc, Dc, DHc, Kc);
    // PvT slice: 64 rows (dh) x 128 elems (k), contiguous (stride Kc).
    tdm_load_2d_bf16(
        (unsigned int)(size_t)&lds_pv[0],
        (unsigned long long)(size_t)(PvT + ((size_t)b * Dc + h * DHc) * Kc),
        Kc, DHc, Kc, Kc, DHc);
    __builtin_amdgcn_s_wait_tensorcnt(0);
  }

  // --- load q fragments while DMA is in flight ------------------------------
  v16bf qa[2];
  {
    const float* qrow = q + (((size_t)b * Hc + h) * Nc + (n0 + lm)) * DHc;
#pragma unroll
    for (int c = 0; c < 2; ++c) {
      const int off = c * 32 + half * 16;
#pragma unroll
      for (int e = 0; e < 8; ++e) {
        qa[c][e]     = (__bf16)qrow[off + e];
        qa[c][e + 8] = (__bf16)qrow[off + 8 + e];
      }
    }
  }
  __syncthreads();   // LDS slices valid for all waves

  // --- dots = q @ kk^T : 8 column tiles of 16 k's ---------------------------
  v8f dots[8];
#pragma unroll
  for (int kt = 0; kt < 8; ++kt) {
    v8f c = {};
#pragma unroll
    for (int ch = 0; ch < 2; ++ch) {
      // B(r=dh_local, col=k) = lds_pk[kt*16 + k][ch*32 + r]
      const v16bf bb =
          *(const v16bf*)&lds_pk[(kt * 16 + lm) * DHc + ch * 32 + half * 16];
      c = WMMA_BF16(qa[ch], bb, c);
    }
    dots[kt] = c;
  }

  // --- softmax over 128 cols per row ---------------------------------------
  const float scale = 0.125f;   // 1/sqrt(64)
#pragma unroll
  for (int i = 0; i < 8; ++i) {
    float mx = -3.0e38f;
#pragma unroll
    for (int kt = 0; kt < 8; ++kt) {
      dots[kt][i] *= scale;
      mx = fmaxf(mx, dots[kt][i]);
    }
#pragma unroll
    for (int mask = 1; mask < 16; mask <<= 1)
      mx = fmaxf(mx, __shfl_xor(mx, mask, 32));
    float s = 0.0f;
#pragma unroll
    for (int kt = 0; kt < 8; ++kt) {
      const float e = __expf(dots[kt][i] - mx);
      dots[kt][i] = e;
      s += e;
    }
#pragma unroll
    for (int mask = 1; mask < 16; mask <<= 1)
      s += __shfl_xor(s, mask, 32);
    const float inv = 1.0f / s;
#pragma unroll
    for (int kt = 0; kt < 8; ++kt) dots[kt][i] *= inv;
  }

  // --- stage attn to LDS (bf16), C-layout -> row-major [16][128] ------------
  __bf16* at = &lds_attn[wave][0];
#pragma unroll
  for (int kt = 0; kt < 8; ++kt)
#pragma unroll
    for (int i = 0; i < 8; ++i) {
      const int row = i + half * 8;
      const int col = kt * 16 + lm;
      at[row * Kc + col] = (__bf16)dots[kt][i];
    }
  __syncthreads();

  // --- out = attn @ vv : 4 dh tiles, reduction over K in 4 chunks of 32 -----
  v8f oacc[4] = {};
#pragma unroll
  for (int ck = 0; ck < 4; ++ck) {
    union { v16bf v; v8bf hlf[2]; } afr;
    const int base = ck * 32 + half * 16;
    afr.hlf[0] = *(const v8bf*)&at[lm * Kc + base];
    afr.hlf[1] = *(const v8bf*)&at[lm * Kc + base + 8];
#pragma unroll
    for (int dt = 0; dt < 4; ++dt) {
      // B(r=k_local, col=dh) = lds_pv[dt*16 + col][ck*32 + r]
      const v16bf bb =
          *(const v16bf*)&lds_pv[(dt * 16 + lm) * Kc + ck * 32 + half * 16];
      oacc[dt] = WMMA_BF16(afr.v, bb, oacc[dt]);
    }
  }

  // --- store f32 output [B, N, D] -------------------------------------------
#pragma unroll
  for (int dt = 0; dt < 4; ++dt)
#pragma unroll
    for (int i = 0; i < 8; ++i) {
      const int m = i + half * 8;
      out[((size_t)b * Nc + n0 + m) * Dc + h * DHc + dt * 16 + lm] = oacc[dt][i];
    }
}

// ---------------------------------------------------------------------------
extern "C" void kernel_launch(void* const* d_in, const int* in_sizes, int n_in,
                              void* d_out, int out_size, void* d_ws, size_t ws_size,
                              hipStream_t stream) {
  (void)in_sizes; (void)n_in; (void)out_size; (void)ws_size;
  const float* queries = (const float*)d_in[0];  // [B,H,N,DH]
  const float* keys    = (const float*)d_in[1];  // [B,N,D]
  const float* values  = (const float*)d_in[2];  // [B,N,D]
  const float* proj_k  = (const float*)d_in[3];  // [N,K]
  const float* proj_v  = (const float*)d_in[4];  // [N,K]

  // Workspace: Pk [B,K,D] bf16 (2 MB) + PvT [B,D,K] bf16 (2 MB)
  __bf16* Pk  = (__bf16*)d_ws;
  __bf16* PvT = Pk + (size_t)Bc * Kc * Dc;

  dim3 g1(Dc / 128, Bc);             // 8 x 8 blocks, 8 waves each
  linformer_proj_kernel<false><<<g1, 256, 0, stream>>>(keys,   proj_k, Pk);
  linformer_proj_kernel<true ><<<g1, 256, 0, stream>>>(values, proj_v, PvT);

  dim3 g2(Nc / 128, Hc, Bc);         // 32 x 16 x 8 blocks, 8 waves each
  linformer_attn_kernel<<<g2, 256, 0, stream>>>(queries, Pk, PvT, (float*)d_out);
}
